// NMSOBBModel_687194767840
// MI455X (gfx1250) — compile-verified, matrix operand order checked by reading, not run on hardware
//
#include <hip/hip_runtime.h>
#include <hip/hip_bf16.h>

typedef __attribute__((ext_vector_type(16))) _Float16 v16h;
typedef __attribute__((ext_vector_type(8)))  float    v8f;

#define MCAND 600
#define MAXDET 300
#define SCORE_THR 0.001f
#define IOU_THR 0.7f
#define EPSF 1e-7f

// ---------------------------------------------------------------------------
// Kernel 0: zero the 4x16 channel-sum accumulator
// ---------------------------------------------------------------------------
__global__ void zero64_kernel(float* __restrict__ p) {
    p[threadIdx.x] = 0.0f;
}

// ---------------------------------------------------------------------------
// Kernel 1: conv3x3(SAME) + bias + relu + maxpool2x2 + full-HW sum reduction.
// One block = one 2-row output band of one batch image.
//   LDS tile: rows 0..11 = 3 channels x 4 input rows (halo, f16), rows 12..15
//   zeroed pad region so K>=27 zero-padding is just an address (branch-free).
//   Element LDS offset = koff(K) [compile-time] + dy*STRIDE + pixel_col.
//   8 waves x 4 WMMAs (V_WMMA_F32_16X16X32_F16): strip = 2x8 conv pixels,
//   N = 16 out channels, K = 27 pad 32; B (weights) register-resident.
//   Vertical 2x2-pool pair = lane^16 swap via v_permlanex16 (VALU, co-execs
//   with WMMA); horizontal pair = adjacent accumulator elements.
// ---------------------------------------------------------------------------
#define TILE_STRIDE 260   // 258 cols + pad

__device__ constexpr int koff(int K) {
    return (K < 27) ? (((K / 9) * 4 + ((K % 9) / 3)) * TILE_STRIDE + (K % 3))
                    : (12 * TILE_STRIDE);   // zeroed pad region
}

__device__ __forceinline__ float xhalf_swap(float v) {
#if __has_builtin(__builtin_amdgcn_permlanex16)
    int vi = __builtin_bit_cast(int, v);
    int oi = __builtin_amdgcn_permlanex16(vi, vi, 0x76543210, 0xfedcba98,
                                          false, false);
    return __builtin_bit_cast(float, oi);
#else
    return __shfl_xor(v, 16, 32);
#endif
}

__global__ __launch_bounds__(256) void conv_pool_reduce_kernel(
    const float* __restrict__ xin, const float* __restrict__ cw,
    const float* __restrict__ cb, float* __restrict__ fsum)
{
    __shared__ _Float16 s_in[16 * TILE_STRIDE];
    __shared__ float    s_acc[16];

    const int tid  = threadIdx.x;
    const int lane = tid & 31;
    const int wavi = tid >> 5;                 // 0..7
    const int b    = blockIdx.x >> 7;          // batch
    const int band = blockIdx.x & 127;
    const int y0   = band << 1;

    if (tid < 16) s_acc[tid] = 0.0f;

    // ---- zero the pad region (rows 12..15) ----
    for (int idx = tid; idx < 4 * TILE_STRIDE; idx += 256)
        s_in[12 * TILE_STRIDE + idx] = (_Float16)0.0f;

    // ---- stage halo tile: rows y0-1..y0+2, cols -1..256, 3 channels ----
    for (int idx = tid; idx < 3 * 4 * 258; idx += 256) {
        const int c  = idx / (4 * 258);
        const int r2 = idx % (4 * 258);
        const int rr = r2 / 258;               // tile row
        const int cc = r2 % 258;               // tile col
        const int gy = y0 + rr - 1;
        const int gx = cc - 1;
        float v = 0.0f;
        if (gy >= 0 && gy < 256 && gx >= 0 && gx < 256)
            v = xin[((b * 3 + c) * 256 + gy) * 256 + gx];
        s_in[(c * 4 + rr) * TILE_STRIDE + cc] = (_Float16)v;
    }

    // ---- per-lane constants ----
    const bool lo = (lane < 16);
    const int  n  = lane & 15;                 // out channel (B / C role)
    const int  m  = lane & 15;                 // A-matrix pixel index
    const int  dy = m >> 3;
    const int  dx = m & 7;

    // A-element base offsets: select between the two compile-time tables
    int base[16];
#pragma unroll
    for (int i = 0; i < 8; ++i) {
        base[i]     = lo ? koff(i)      : koff(8 + i);
        base[8 + i] = lo ? koff(16 + i) : koff(24 + i);
    }

    // ---- weights: register-resident B matrix (clamp+select, branch-free) ----
    v16h Bm;
#pragma unroll
    for (int i = 0; i < 8; ++i) {
        const int K0 = lo ? i : 8 + i;         // <= 15, always valid
        const int K1 = lo ? 16 + i : 24 + i;   // 16..31
        const int Kc = (K1 < 27) ? K1 : 26;
        float w0 = cw[n * 27 + K0];
        float w1 = cw[n * 27 + Kc];
        w1 = (K1 < 27) ? w1 : 0.0f;
        Bm[i]     = (_Float16)w0;
        Bm[8 + i] = (_Float16)w1;
    }
    const float bias = cb[n];

    __syncthreads();

    // ---- 4 strips per wave: inner loop is add + 16 ds_load_u16 + wmma ----
    float chsum = 0.0f;
#pragma unroll
    for (int it = 0; it < 4; ++it) {
        const int common = dy * TILE_STRIDE + (((wavi << 2) + it) << 3) + dx;

        v16h A;
#pragma unroll
        for (int e = 0; e < 16; ++e)
            A[e] = s_in[base[e] + common];

        v8f acc = {0.f, 0.f, 0.f, 0.f, 0.f, 0.f, 0.f, 0.f};
        acc = __builtin_amdgcn_wmma_f32_16x16x32_f16(
            /*neg_a=*/false, A, /*neg_b=*/false, Bm,
            /*c_mod=*/(short)0, acc, /*reuse_a=*/false, /*reuse_b=*/false);

        float vm[8];
#pragma unroll
        for (int j = 0; j < 8; ++j) {
            float v = fmaxf(acc[j] + bias, 0.0f);   // bias + relu
            vm[j] = fmaxf(v, xhalf_swap(v));        // vertical pool pair
        }
#pragma unroll
        for (int k = 0; k < 4; ++k)                 // horizontal pool + sum
            chsum += fmaxf(vm[2 * k], vm[2 * k + 1]);
    }

    if (lane < 16) atomicAdd(&s_acc[n], chsum);     // whole block same batch
    __syncthreads();
    if (tid < 16) atomicAdd(&fsum[b * 16 + tid], s_acc[tid]);
}

// ---------------------------------------------------------------------------
// Kernel 2: heads. fmean = fsum/16384; boxes/logits/angle are 16-wide dots.
// Emits candidate boxes5 (q = d*2+k shares det box) and softmax scores.
// ---------------------------------------------------------------------------
__global__ __launch_bounds__(256) void heads_kernel(
    const float* __restrict__ fsum,
    const float* __restrict__ bw, const float* __restrict__ bb,
    const float* __restrict__ cw, const float* __restrict__ cbv,
    const float* __restrict__ aw, const float* __restrict__ ab,
    float* __restrict__ cand_box, float* __restrict__ cand_s)
{
    __shared__ float fm[16];
    const int b = blockIdx.x, tid = threadIdx.x;
    if (tid < 16) fm[tid] = fsum[b * 16 + tid] * (1.0f / 16384.0f);
    __syncthreads();

    for (int d = tid; d < MAXDET; d += 256) {
        float bx[5];
#pragma unroll
        for (int j = 0; j < 4; ++j) {
            const float* w = bw + (d * 4 + j) * 16;
            float s = bb[d * 4 + j];
#pragma unroll
            for (int c = 0; c < 16; ++c) s += w[c] * fm[c];
            bx[j] = s;
        }
        {
            const float* w = aw + d * 16;
            float s = ab[d];
#pragma unroll
            for (int c = 0; c < 16; ++c) s += w[c] * fm[c];
            bx[4] = s;
        }
        float lg[2];
#pragma unroll
        for (int k = 0; k < 2; ++k) {
            const float* w = cw + (d * 2 + k) * 16;
            float s = cbv[d * 2 + k];
#pragma unroll
            for (int c = 0; c < 16; ++c) s += w[c] * fm[c];
            lg[k] = s;
        }
        const float mx = fmaxf(lg[0], lg[1]);
        const float e0 = expf(lg[0] - mx), e1 = expf(lg[1] - mx);
        const float inv = 1.0f / (e0 + e1);
        const int base = b * MCAND + d * 2;
#pragma unroll
        for (int k = 0; k < 2; ++k) {
            float* o = cand_box + (base + k) * 5;
#pragma unroll
            for (int j = 0; j < 5; ++j) o[j] = bx[j];
        }
        cand_s[base]     = e0 * inv;
        cand_s[base + 1] = e1 * inv;
    }
}

// ---------------------------------------------------------------------------
// Kernel 3: per-batch NMS. Bitonic sort (1024 padded) in LDS, probiou greedy
// suppression (one barrier per sequential step), compaction + padding.
// ---------------------------------------------------------------------------
__global__ __launch_bounds__(256) void nms_kernel(
    const float* __restrict__ cand_box, const float* __restrict__ cand_s,
    float* __restrict__ out)
{
    __shared__ float key[1024];
    __shared__ int   sidx[1024];
    __shared__ float sbox[MCAND * 5];   // sorted (cx,cy,w,h,t) for output
    __shared__ float gpar[MCAND * 5];   // (x,y,a,b,c) gaussian params
    __shared__ float ssc[MCAND];
    __shared__ int   slb[MCAND];
    __shared__ int   skeep[MCAND];

    const int b = blockIdx.x, tid = threadIdx.x;

    for (int i = tid; i < 1024; i += 256) {
        key[i]  = (i < MCAND) ? cand_s[b * MCAND + i] : -1e30f;
        sidx[i] = i;
    }
    __syncthreads();

    // bitonic sort, descending by score, ties -> lower original index first
    for (int k = 2; k <= 1024; k <<= 1) {
        for (int j = k >> 1; j > 0; j >>= 1) {
#pragma unroll
            for (int p = 0; p < 4; ++p) {
                const int i = tid + p * 256;
                const int l = i ^ j;
                if (l > i) {
                    const float ki = key[i], kl = key[l];
                    const int   ii = sidx[i], il = sidx[l];
                    const bool iBeforeL = (ki > kl) || (ki == kl && ii < il);
                    const bool desc = ((i & k) == 0);
                    if (desc ? !iBeforeL : iBeforeL) {
                        key[i] = kl;  key[l] = ki;
                        sidx[i] = il; sidx[l] = ii;
                    }
                }
            }
            __syncthreads();
        }
    }

    // decode sorted candidates -> gaussian params
    for (int i = tid; i < MCAND; i += 256) {
        const int q = sidx[i];
        const float* bp = cand_box + (b * MCAND + q) * 5;
        const float cx = bp[0], cy = bp[1], w = bp[2], h = bp[3], t = bp[4];
        const float ct = cosf(t), st = sinf(t);
        const float w2 = w * w * (1.0f / 12.0f), h2 = h * h * (1.0f / 12.0f);
        sbox[i * 5 + 0] = cx; sbox[i * 5 + 1] = cy; sbox[i * 5 + 2] = w;
        sbox[i * 5 + 3] = h;  sbox[i * 5 + 4] = t;
        gpar[i * 5 + 0] = cx;
        gpar[i * 5 + 1] = cy;
        gpar[i * 5 + 2] = w2 * ct * ct + h2 * st * st;
        gpar[i * 5 + 3] = w2 * st * st + h2 * ct * ct;
        gpar[i * 5 + 4] = (w2 - h2) * ct * st;
        ssc[i]   = key[i];
        slb[i]   = q & 1;
        skeep[i] = (key[i] > SCORE_THR) ? 1 : 0;
    }
    __syncthreads();

    // sequential greedy suppression
    for (int i = 0; i < MCAND; ++i) {
        if (skeep[i]) {
            const float x1 = gpar[i * 5 + 0], y1 = gpar[i * 5 + 1];
            const float a1 = gpar[i * 5 + 2], b1 = gpar[i * 5 + 3];
            const float c1 = gpar[i * 5 + 4];
            const int   li = slb[i];
            for (int jj = i + 1 + tid; jj < MCAND; jj += 256) {
                if (skeep[jj] && slb[jj] == li) {
                    const float x2 = gpar[jj * 5 + 0], y2 = gpar[jj * 5 + 1];
                    const float a2 = gpar[jj * 5 + 2], b2 = gpar[jj * 5 + 3];
                    const float c2 = gpar[jj * 5 + 4];
                    const float As = a1 + a2, Bs = b1 + b2, Cs = c1 + c2;
                    const float denom = As * Bs - Cs * Cs + EPSF;
                    const float dxv = x1 - x2, dyv = y1 - y2;
                    const float t1 = 0.25f * (As * dyv * dyv + Bs * dxv * dxv) / denom;
                    const float t2 = 0.5f * (Cs * (x2 - x1) * (y1 - y2)) / denom;
                    const float d1 = fmaxf(a1 * b1 - c1 * c1, 0.0f);
                    const float d2 = fmaxf(a2 * b2 - c2 * c2, 0.0f);
                    const float t3 = 0.5f * logf(denom / (4.0f * sqrtf(d1 * d2) + EPSF) + EPSF);
                    const float bd = fminf(fmaxf(t1 + t2 + t3, EPSF), 100.0f);
                    const float hd = sqrtf(1.0f - expf(-bd) + EPSF);
                    if ((1.0f - hd) > IOU_THR) skeep[jj] = 0;
                }
            }
        }
        __syncthreads();
    }

    // compaction + zero padding + n_valid
    if (tid == 0) {
        float* ob = out + b * (MAXDET * 5);
        float* os = out + 4 * MAXDET * 5 + b * MAXDET;
        float* ol = out + 4 * MAXDET * 5 + 4 * MAXDET + b * MAXDET;
        int cnt = 0;
        for (int i = 0; i < MCAND; ++i) {
            if (skeep[i]) {
                if (cnt < MAXDET) {
                    for (int j = 0; j < 5; ++j) ob[cnt * 5 + j] = sbox[i * 5 + j];
                    os[cnt] = ssc[i];
                    ol[cnt] = (float)slb[i];
                }
                cnt++;
            }
        }
        const int nv = (cnt < MAXDET) ? cnt : MAXDET;
        for (int s = nv; s < MAXDET; ++s) {
            for (int j = 0; j < 5; ++j) ob[s * 5 + j] = 0.0f;
            os[s] = 0.0f;
            ol[s] = 0.0f;
        }
        out[4 * MAXDET * 5 + 8 * MAXDET + b] = (float)nv;
    }
}

// ---------------------------------------------------------------------------
extern "C" void kernel_launch(void* const* d_in, const int* in_sizes, int n_in,
                              void* d_out, int out_size, void* d_ws, size_t ws_size,
                              hipStream_t stream)
{
    const float* x      = (const float*)d_in[0];
    const float* conv_w = (const float*)d_in[1];
    const float* conv_b = (const float*)d_in[2];
    const float* bbox_w = (const float*)d_in[3];
    const float* bbox_b = (const float*)d_in[4];
    const float* cls_w  = (const float*)d_in[5];
    const float* cls_b  = (const float*)d_in[6];
    const float* ang_w  = (const float*)d_in[7];
    const float* ang_b  = (const float*)d_in[8];
    float* out = (float*)d_out;

    // workspace layout (floats): fsum[64] | cand_box[4*600*5] | cand_s[4*600]
    float* fsum     = (float*)d_ws;
    float* cand_box = fsum + 64;
    float* cand_s   = cand_box + 4 * MCAND * 5;

    zero64_kernel<<<1, 64, 0, stream>>>(fsum);
    // 512 blocks = 4 batches * 128 bands; 8 waves/block * 4 WMMA strips/wave
    conv_pool_reduce_kernel<<<512, 256, 0, stream>>>(x, conv_w, conv_b, fsum);
    heads_kernel<<<4, 256, 0, stream>>>(fsum, bbox_w, bbox_b, cls_w, cls_b,
                                        ang_w, ang_b, cand_box, cand_s);
    nms_kernel<<<4, 256, 0, stream>>>(cand_box, cand_s, out);
}
